// VectorQuantizer_9887014716117
// MI455X (gfx1250) — matrix operand order repaired and checked by reading, not compiled, not run on hardware
//
#include <hip/hip_runtime.h>

// ---------------------------------------------------------------------------
// VQ-VAE vector quantizer for MI455X (gfx1250, wave32, WMMA).
// d2[n,k] = |x_n|^2 - 2 x_n·c_k + |c_k|^2 ; argmin_k invariant to |x_n|^2.
// Score GEMM done with v_wmma_f32_16x16x32_bf16 using bf16 hi/lo splitting
// (3 WMMAs per tile) so the argmin tracks the fp32 reference.
// ---------------------------------------------------------------------------

typedef __attribute__((ext_vector_type(16))) __bf16 v16bf;
typedef __attribute__((ext_vector_type(8)))  __bf16 v8bf;
typedef __attribute__((ext_vector_type(8)))  float  v8f;

#define VQ_D        256      // feature dim (fixed by reference)
#define CHUNK_ROWS  64       // codebook rows staged per LDS chunk
#define LDS_STRIDE  264      // 256 + 8 bf16 pad -> conflict-free b128 LDS reads

#if __has_builtin(__builtin_amdgcn_global_load_async_to_lds_b128) && \
    __has_builtin(__builtin_amdgcn_s_wait_asynccnt)
#define VQ_ASYNC 1
#else
#define VQ_ASYNC 0
#endif

#if VQ_ASYNC
// Builtin signature (from hipcc diagnostic): first param is
// 'int __vector(4) __device__ *' i.e. addrspace(1) pointer to 16-byte vector.
typedef int vq_v4i __attribute__((vector_size(16)));
typedef __attribute__((address_space(1))) vq_v4i vq_gv4i;
typedef __attribute__((address_space(3))) vq_v4i vq_lv4i;

__device__ __forceinline__ void vq_async_cp16(const void* g, void* l) {
  // 16 bytes per lane, global -> LDS, tracked by ASYNCcnt.
  // AS pointers fabricated via integer round-trip (AS3 flat low 32 bits ==
  // LDS offset on amdgcn).
  __builtin_amdgcn_global_load_async_to_lds_b128(
      (vq_gv4i*)(unsigned long long)g,
      (vq_lv4i*)(unsigned int)(unsigned long long)l, 0, 0);
}
#endif

// ---------------------------------------------------------------------------
// Kernel 1: split codebook into bf16 hi/lo and compute c_sq in fp32.
// grid = K blocks, 256 threads (one block per codebook row).
// ---------------------------------------------------------------------------
__global__ __launch_bounds__(256)
void vq_prep_kernel(const float* __restrict__ cb,
                    __bf16* __restrict__ cb_hi, __bf16* __restrict__ cb_lo,
                    float* __restrict__ c_sq) {
  const int row = blockIdx.x;
  const int t   = threadIdx.x;
  const size_t i = (size_t)row * VQ_D + t;
  const float v = cb[i];
  const __bf16 h = (__bf16)v;
  cb_hi[i] = h;
  cb_lo[i] = (__bf16)(v - (float)h);
  float s = v * v;
#pragma unroll
  for (int off = 16; off > 0; off >>= 1) s += __shfl_xor(s, off, 32);
  __shared__ float red[8];
  if ((t & 31) == 0) red[t >> 5] = s;
  __syncthreads();
  if (t == 0) {
    float tot = 0.f;
#pragma unroll
    for (int j = 0; j < 8; ++j) tot += red[j];
    c_sq[row] = tot;
  }
}

// ---------------------------------------------------------------------------
// Kernel 2: per-row argmin over K via WMMA bf16x3.
// 256 threads (8 waves). Wave w owns rows [block*128 + w*16, +16).
// Codebook staged in LDS 64 rows at a time, double-buffered.
// ---------------------------------------------------------------------------
__global__ __launch_bounds__(256)
void vq_argmin_kernel(const float* __restrict__ x,
                      const __bf16* __restrict__ cb_hi,
                      const __bf16* __restrict__ cb_lo,
                      const float* __restrict__ c_sq,
                      int* __restrict__ indices, int K) {
  // [buffer][hi=0/lo=1][row*LDS_STRIDE + col]
  __shared__ __align__(16) __bf16 lds_buf[2][2][CHUNK_ROWS * LDS_STRIDE];

  const int t    = threadIdx.x;
  const int lane = t & 31;
  const int wave = t >> 5;
  const int ln   = lane & 15;   // low lane id
  const int hf   = lane >> 4;   // half-wave select
  const int m_base = blockIdx.x * 128 + wave * 16;
  const int row    = m_base + ln;

  // ---- A fragments (x rows), bf16 hi/lo, resident in VGPRs ----------------
  // 16-bit A 16x32 layout (ISA 7.12.2): lane (ln,hf) elements
  //   e in [0,8)  -> K = 32*d + 8*hf + e
  //   e in [8,16) -> K = 32*d + 16 + 8*hf + (e-8)
  v16bf afh[8], afl[8];
  {
    const float* xr = x + (size_t)row * VQ_D;
#pragma unroll
    for (int d = 0; d < 8; ++d) {
      const float4* p0 = (const float4*)(xr + d * 32 + hf * 8);
      const float4* p1 = (const float4*)(xr + d * 32 + 16 + hf * 8);
      float f[16];
      float4 a0 = p0[0], a1 = p0[1], b0 = p1[0], b1 = p1[1];
      f[0]=a0.x; f[1]=a0.y; f[2]=a0.z; f[3]=a0.w;
      f[4]=a1.x; f[5]=a1.y; f[6]=a1.z; f[7]=a1.w;
      f[8]=b0.x; f[9]=b0.y; f[10]=b0.z; f[11]=b0.w;
      f[12]=b1.x; f[13]=b1.y; f[14]=b1.z; f[15]=b1.w;
      v16bf vh, vl;
#pragma unroll
      for (int e = 0; e < 16; ++e) {
        const __bf16 h = (__bf16)f[e];
        vh[e] = h;
        vl[e] = (__bf16)(f[e] - (float)h);
      }
      afh[d] = vh; afl[d] = vl;
    }
  }

  float bestval[8];
  int   bestidx[8];
#pragma unroll
  for (int v = 0; v < 8; ++v) { bestval[v] = 3.402823466e38f; bestidx[v] = 0; }

  // ---- staging assignment: thread t copies 64 bf16 per (hi,lo) buffer -----
  const int sr = t >> 2;   // staged row   0..63
  const int sq = t & 3;    // quarter      0..3 (64 cols each)
  const int nch = K / CHUNK_ROWS;

  auto stage_chunk = [&](int kchunk, int buf) {
    const size_t goff = (size_t)(kchunk * CHUNK_ROWS + sr) * VQ_D + sq * 64;
    const int    loff = sr * LDS_STRIDE + sq * 64;
#if VQ_ASYNC
    const __bf16* gh = cb_hi + goff;
    const __bf16* gl = cb_lo + goff;
    __bf16* lh = &lds_buf[buf][0][loff];
    __bf16* ll = &lds_buf[buf][1][loff];
#pragma unroll
    for (int i = 0; i < 8; ++i) vq_async_cp16(gh + i * 8, lh + i * 8);
#pragma unroll
    for (int i = 0; i < 8; ++i) vq_async_cp16(gl + i * 8, ll + i * 8);
#else
    const uint4* gh = (const uint4*)(cb_hi + goff);
    const uint4* gl = (const uint4*)(cb_lo + goff);
    uint4* lh = (uint4*)&lds_buf[buf][0][loff];
    uint4* ll = (uint4*)&lds_buf[buf][1][loff];
    uint4 th[8], tl[8];
#pragma unroll
    for (int i = 0; i < 8; ++i) { th[i] = gh[i]; tl[i] = gl[i]; }
#pragma unroll
    for (int i = 0; i < 8; ++i) { lh[i] = th[i]; ll[i] = tl[i]; }
#endif
  };

  stage_chunk(0, 0);

  for (int kc = 0; kc < nch; ++kc) {
    const int cur = kc & 1;
#if VQ_ASYNC
    __builtin_amdgcn_s_wait_asynccnt(0);
#endif
    __syncthreads();                       // chunk kc fully in LDS; prev compute done
    if (kc + 1 < nch) stage_chunk(kc + 1, cur ^ 1);

    const __bf16* Lh = lds_buf[cur][0];
    const __bf16* Ll = lds_buf[cur][1];

#pragma unroll
    for (int c = 0; c < 4; ++c) {
      v8f acc = {0.f, 0.f, 0.f, 0.f, 0.f, 0.f, 0.f, 0.f};
      const int col = c * 16 + ln;
      // 16-bit B 32x16 layout: lane (ln,hf) e<8 -> K = 32*d + 8*hf + e, e>=8 -> K = 32*d + 16 + 8*hf + (e-8)
      const __bf16* bh = Lh + col * LDS_STRIDE + hf * 8;
      const __bf16* bl = Ll + col * LDS_STRIDE + hf * 8;
#pragma unroll
      for (int d = 0; d < 8; ++d) {
        union { v16bf v; v8bf h[2]; } Bh, Bl;
        Bh.h[0] = *(const v8bf*)(bh + d * 32);
        Bh.h[1] = *(const v8bf*)(bh + d * 32 + 16);
        Bl.h[0] = *(const v8bf*)(bl + d * 32);
        Bl.h[1] = *(const v8bf*)(bl + d * 32 + 16);
        // small terms first, then the dominant hi*hi term
        acc = __builtin_amdgcn_wmma_f32_16x16x32_bf16(false, afl[d], false, Bh.v,
                                                      (short)0, acc, false, false);
        acc = __builtin_amdgcn_wmma_f32_16x16x32_bf16(false, afh[d], false, Bl.v,
                                                      (short)0, acc, false, false);
        acc = __builtin_amdgcn_wmma_f32_16x16x32_bf16(false, afh[d], false, Bh.v,
                                                      (short)0, acc, false, false);
      }
      const int kglob = kc * CHUNK_ROWS + col;
      const float csq = c_sq[kglob];
      // D layout: acc[v] at this lane is row M = v + 8*hf, col N = ln.
#pragma unroll
      for (int v = 0; v < 8; ++v) {
        const float dval = __builtin_fmaf(-2.0f, acc[v], csq);
        if (dval < bestval[v]) { bestval[v] = dval; bestidx[v] = kglob; }
      }
    }
  }

  // ---- reduce across the 16 lanes holding each row, tie-break lowest k ----
#pragma unroll
  for (int v = 0; v < 8; ++v) {
    float bv = bestval[v];
    int   bi = bestidx[v];
#pragma unroll
    for (int off = 1; off < 16; off <<= 1) {
      const float ov = __shfl_xor(bv, off, 32);
      const int   oi = __shfl_xor(bi, off, 32);
      if (ov < bv || (ov == bv && oi < bi)) { bv = ov; bi = oi; }
    }
    if (ln == 0) indices[m_base + v + 8 * hf] = bi;
  }
}

// ---------------------------------------------------------------------------
// Kernel 3: gather quantized rows (fp32 codebook) + partial loss sums.
// 256 threads, 32 rows per block; deterministic reduction.
// ---------------------------------------------------------------------------
__global__ __launch_bounds__(256)
void vq_gather_loss_kernel(const float* __restrict__ x,
                           const float* __restrict__ cb,
                           const int* __restrict__ idxbuf,
                           float* __restrict__ out,
                           float* __restrict__ partials) {
  const int t = threadIdx.x;
  const int base = blockIdx.x * 32;
  float s = 0.f;
  for (int r = 0; r < 32; ++r) {
    const int n = base + r;
    const int k = idxbuf[n];
    const float q  = cb[(size_t)k * VQ_D + t];
    const float xv = x[(size_t)n * VQ_D + t];
    out[(size_t)n * VQ_D + t] = q;
    const float d = q - xv;
    s = __builtin_fmaf(d, d, s);
  }
#pragma unroll
  for (int off = 16; off > 0; off >>= 1) s += __shfl_xor(s, off, 32);
  __shared__ float red[8];
  if ((t & 31) == 0) red[t >> 5] = s;
  __syncthreads();
  if (t == 0) {
    float tot = 0.f;
#pragma unroll
    for (int j = 0; j < 8; ++j) tot += red[j];
    partials[blockIdx.x] = tot;
  }
}

__global__ __launch_bounds__(256)
void vq_finalize_kernel(const float* __restrict__ partials, int n, float scale,
                        float* __restrict__ loss_out) {
  const int t = threadIdx.x;
  float s = 0.f;
  for (int i = t; i < n; i += 256) s += partials[i];
#pragma unroll
  for (int off = 16; off > 0; off >>= 1) s += __shfl_xor(s, off, 32);
  __shared__ float red[8];
  if ((t & 31) == 0) red[t >> 5] = s;
  __syncthreads();
  if (t == 0) {
    float tot = 0.f;
#pragma unroll
    for (int j = 0; j < 8; ++j) tot += red[j];
    *loss_out = tot * scale;   // (1 + 0.25) * mean
  }
}

// ---------------------------------------------------------------------------
extern "C" void kernel_launch(void* const* d_in, const int* in_sizes, int n_in,
                              void* d_out, int out_size, void* d_ws, size_t ws_size,
                              hipStream_t stream) {
  const float* x  = (const float*)d_in[0];
  const float* cb = (const float*)d_in[1];
  const int N = in_sizes[0] / VQ_D;     // 32768
  const int K = in_sizes[1] / VQ_D;     // 4096
  float* out = (float*)d_out;           // [N*D] quantized, then [1] loss

  // workspace layout
  unsigned char* w = (unsigned char*)d_ws;
  __bf16* cb_hi = (__bf16*)w;                                   // K*D bf16
  __bf16* cb_lo = (__bf16*)(w + (size_t)K * VQ_D * 2);          // K*D bf16
  float*  c_sq  = (float*)(w + (size_t)K * VQ_D * 4);           // K f32
  int*    idxb  = (int*)((unsigned char*)c_sq + (size_t)K * 4); // N i32
  float*  parts = (float*)((unsigned char*)idxb + (size_t)N * 4);

  const int ngather = N / 32;

  vq_prep_kernel<<<K, 256, 0, stream>>>(cb, cb_hi, cb_lo, c_sq);
  vq_argmin_kernel<<<N / 128, 256, 0, stream>>>(x, cb_hi, cb_lo, c_sq, idxb, K);
  vq_gather_loss_kernel<<<ngather, 256, 0, stream>>>(x, cb, idxb, out, parts);
  vq_finalize_kernel<<<1, 256, 0, stream>>>(parts, ngather,
                                            1.25f / ((float)N * (float)VQ_D),
                                            out + (size_t)N * VQ_D);
}
